// VQVAE_86414741995866
// MI455X (gfx1250) — matrix operand order repaired and checked by reading, not compiled, not run on hardware
//
#include <hip/hip_runtime.h>
#include <cstddef>

// ============================ device types =============================
typedef __attribute__((ext_vector_type(16))) __bf16        v16bf;
typedef __attribute__((ext_vector_type(8)))  float         v8f;
typedef __attribute__((ext_vector_type(8)))  unsigned int  v8u;

__device__ __forceinline__ int imin(int a, int b) { return a < b ? a : b; }

// pack two f32 -> bf16x2 (round-to-nearest, ties away): cheap VALU form
__device__ __forceinline__ unsigned int pack2_bf16(float a, float b) {
  unsigned int ua = __builtin_bit_cast(unsigned int, a) + 0x8000u;
  unsigned int ub = __builtin_bit_cast(unsigned int, b) + 0x8000u;
  return (ua >> 16) | (ub & 0xFFFF0000u);
}

// ===================== WMMA GEMM: C = act(A@W + b [+ R]) ===============
// A: M x K fp32 row-major (or transposed mode, see below),
// W: K x N fp32 row-major, C: M x N fp32.
// bf16 compute, fp32 accumulate via v_wmma_f32_16x16x32_bf16 (wave32).
// 256 threads = 8 waves; wave (wr 0..3, wc 0..1) owns a 32x64 patch
// of the 128x128 block tile -> 2x4 16x16 accumulators, 8 WMMA / K-step.
//
// aTransD != 0 enables transposed-A mode: logical A row rl=(b*D+d), k=t,
// element at Aglob[((rl/D)*K + k)*D + (rl%D)].  BM=128 divides D, so
// b and d0 are uniform per block: base + k*D + r, coalesced along r.
#define BM 128
#define BN 128
#define BK 32
#define LDA 40   // padded bf16 row stride for A tile (rows of K)
#define LDB 40   // padded bf16 row stride for transposed B tile (rows of N)

// ---- tile staging helpers: always-valid clamped loads, no EXEC branching ----
__device__ __forceinline__ void gload_A(const float* __restrict__ A, int M, int K,
                                        int rowBlk, int kb, int tid, float va[16]) {
  if (rowBlk + BM <= M && kb + BK <= K) {
#pragma unroll
    for (int i = 0; i < 8; ++i) {
      int idx = tid + i * 256;                 // pair index over 128r x 16 k-pairs
      int r = idx >> 4, c = (idx & 15) * 2;
      const float* p = &A[(size_t)(rowBlk + r) * K + kb + c];
      va[2 * i] = p[0]; va[2 * i + 1] = p[1];
    }
  } else {
#pragma unroll
    for (int i = 0; i < 8; ++i) {
      int idx = tid + i * 256;
      int r = idx >> 4, c = (idx & 15) * 2;
      int gr = rowBlk + r, gc0 = kb + c, gc1 = gc0 + 1;
      int cr = imin(gr, M - 1);
      float v0 = A[(size_t)cr * K + imin(gc0, K - 1)];
      float v1 = A[(size_t)cr * K + imin(gc1, K - 1)];
      va[2 * i]     = (gr < M && gc0 < K) ? v0 : 0.f;
      va[2 * i + 1] = (gr < M && gc1 < K) ? v1 : 0.f;
    }
  }
}

__device__ __forceinline__ void sstore_A(unsigned short* Ash, int tid, const float va[16]) {
#pragma unroll
  for (int i = 0; i < 8; ++i) {
    int idx = tid + i * 256;
    int r = idx >> 4, c = (idx & 15) * 2;
    *(unsigned int*)&Ash[r * LDA + c] = pack2_bf16(va[2 * i], va[2 * i + 1]);
  }
}

// transposed-A loader: 128 r (contiguous in memory) x 16 k-pairs
__device__ __forceinline__ void gload_AT(const float* __restrict__ A, int K, int D,
                                         int rowBlk, int kb, int tid, float va[16]) {
  const size_t base = (size_t)(rowBlk / D) * K * D + (size_t)(rowBlk % D);
  if (kb + BK <= K) {
#pragma unroll
    for (int i = 0; i < 8; ++i) {
      int idx = tid + i * 256;
      int r = idx & 127, k = (idx >> 7) * 2;
      va[2 * i]     = A[base + (size_t)(kb + k) * D + r];
      va[2 * i + 1] = A[base + (size_t)(kb + k + 1) * D + r];
    }
  } else {
#pragma unroll
    for (int i = 0; i < 8; ++i) {
      int idx = tid + i * 256;
      int r = idx & 127, k = (idx >> 7) * 2;
      int gk0 = kb + k, gk1 = gk0 + 1;
      float v0 = A[base + (size_t)imin(gk0, K - 1) * D + r];
      float v1 = A[base + (size_t)imin(gk1, K - 1) * D + r];
      va[2 * i]     = (gk0 < K) ? v0 : 0.f;
      va[2 * i + 1] = (gk1 < K) ? v1 : 0.f;
    }
  }
}

__device__ __forceinline__ void sstore_AT(unsigned short* Ash, int tid, const float va[16]) {
#pragma unroll
  for (int i = 0; i < 8; ++i) {
    int idx = tid + i * 256;
    int r = idx & 127, k = (idx >> 7) * 2;
    *(unsigned int*)&Ash[r * LDA + k] = pack2_bf16(va[2 * i], va[2 * i + 1]);
  }
}

__device__ __forceinline__ void gload_B(const float* __restrict__ W, int K, int N,
                                        int kb, int colBlk, int tid, float vb[16]) {
  if (colBlk + BN <= N && kb + BK <= K) {
#pragma unroll
    for (int i = 0; i < 8; ++i) {
      int idx = tid + i * 256;                 // 128 n x 16 k-pairs
      int n = idx & 127, k = (idx >> 7) * 2;
      vb[2 * i]     = W[(size_t)(kb + k) * N + colBlk + n];
      vb[2 * i + 1] = W[(size_t)(kb + k + 1) * N + colBlk + n];
    }
  } else {
#pragma unroll
    for (int i = 0; i < 8; ++i) {
      int idx = tid + i * 256;
      int n = idx & 127, k = (idx >> 7) * 2;
      int gn = colBlk + n, gk0 = kb + k, gk1 = gk0 + 1;
      int cn = imin(gn, N - 1);
      float v0 = W[(size_t)imin(gk0, K - 1) * N + cn];
      float v1 = W[(size_t)imin(gk1, K - 1) * N + cn];
      vb[2 * i]     = (gn < N && gk0 < K) ? v0 : 0.f;
      vb[2 * i + 1] = (gn < N && gk1 < K) ? v1 : 0.f;
    }
  }
}

__device__ __forceinline__ void sstore_B(unsigned short* Bsh, int tid, const float vb[16]) {
#pragma unroll
  for (int i = 0; i < 8; ++i) {
    int idx = tid + i * 256;
    int n = idx & 127, k = (idx >> 7) * 2;
    *(unsigned int*)&Bsh[n * LDB + k] = pack2_bf16(vb[2 * i], vb[2 * i + 1]);
  }
}

__global__ __launch_bounds__(256) void gemm_kernel(
    const float* __restrict__ A, const float* __restrict__ W,
    const float* __restrict__ bias, const float* __restrict__ resid,
    float* __restrict__ C, int M, int N, int K, int act, int aTransD)
{
  __shared__ __align__(16) unsigned short Ash[2][BM * LDA];  // double buffered
  __shared__ __align__(16) unsigned short Bsh[2][BN * LDB];

  const int tid  = threadIdx.x;
  const int lane = tid & 31;
  const int wave = tid >> 5;
  const int wr   = wave >> 1;      // 0..3 : 32-row slab
  const int wc   = wave & 1;       // 0..1 : 64-col slab
  const int m16  = lane & 15;
  const int kgrp = lane >> 4;      // half-wave group (ISA 16-bit A/B layout)

  const int rowBlk = blockIdx.y * BM;
  const int colBlk = blockIdx.x * BN;

  v8f zero = {0.f,0.f,0.f,0.f,0.f,0.f,0.f,0.f};
  v8f acc[2][4];
#pragma unroll
  for (int ti = 0; ti < 2; ++ti)
#pragma unroll
    for (int tj = 0; tj < 4; ++tj) acc[ti][tj] = zero;

  float va[16], vb[16];
  if (aTransD) gload_AT(A, K, aTransD, rowBlk, 0, tid, va);
  else         gload_A (A, M, K, rowBlk, 0, tid, va);
  gload_B(W, K, N, 0, colBlk, tid, vb);
  if (aTransD) sstore_AT(Ash[0], tid, va);
  else         sstore_A (Ash[0], tid, va);
  sstore_B(Bsh[0], tid, vb);
  __syncthreads();

  const int nIter = (K + BK - 1) / BK;
  for (int it = 0; it < nIter; ++it) {
    const unsigned short* cA = Ash[it & 1];
    const unsigned short* cB = Bsh[it & 1];
    unsigned short* nA = Ash[(it + 1) & 1];
    unsigned short* nB = Bsh[(it + 1) & 1];
    const bool more = (it + 1) < nIter;

    if (more) {                      // issue next tile's global loads early
      int kb = (it + 1) * BK;
      if (aTransD) gload_AT(A, K, aTransD, rowBlk, kb, tid, va);
      else         gload_A (A, M, K, rowBlk, kb, tid, va);
      gload_B(W, K, N, kb, colBlk, tid, vb);
      if (it + 2 < nIter) {          // speculative prefetch 2 tiles ahead
        int kp = (it + 2) * BK;
        if (aTransD) {
          size_t base = (size_t)(rowBlk / aTransD) * K * aTransD + (size_t)(rowBlk % aTransD);
          __builtin_prefetch(&A[base + (size_t)imin(kp + (tid >> 7) * 2, K - 1) * aTransD + (tid & 127)], 0, 0);
        } else {
          __builtin_prefetch(&A[(size_t)imin(rowBlk + (tid >> 4), M - 1) * K + imin(kp + (tid & 15) * 2, K - 1)], 0, 0);
        }
        __builtin_prefetch(&W[(size_t)imin(kp + (tid >> 7) * 2, K - 1) * N + imin(colBlk + (tid & 127), N - 1)], 0, 0);
      }
    }

    // ---- fragments per ISA 16-bit 16x32 layout ----
    v16bf aF[2], bF[4];
#pragma unroll
    for (int ti = 0; ti < 2; ++ti) {
      int r = wr * 32 + ti * 16 + m16;
      uint4 lo = *(const uint4*)&cA[r * LDA + kgrp * 8];       // K g*8..g*8+7
      uint4 hi = *(const uint4*)&cA[r * LDA + kgrp * 8 + 16];  // K g*8+16..+23
      v8u u;
      u[0] = lo.x; u[1] = lo.y; u[2] = lo.z; u[3] = lo.w;
      u[4] = hi.x; u[5] = hi.y; u[6] = hi.z; u[7] = hi.w;
      aF[ti] = __builtin_bit_cast(v16bf, u);
    }
#pragma unroll
    for (int tj = 0; tj < 4; ++tj) {
      int n = wc * 64 + tj * 16 + m16;
      uint4 lo = *(const uint4*)&cB[n * LDB + kgrp * 8];       // K g*8..g*8+7
      uint4 hi = *(const uint4*)&cB[n * LDB + kgrp * 8 + 16];  // K g*8+16..+23
      v8u u;
      u[0] = lo.x; u[1] = lo.y; u[2] = lo.z; u[3] = lo.w;
      u[4] = hi.x; u[5] = hi.y; u[6] = hi.z; u[7] = hi.w;
      bF[tj] = __builtin_bit_cast(v16bf, u);
    }

#pragma unroll
    for (int ti = 0; ti < 2; ++ti)
#pragma unroll
      for (int tj = 0; tj < 4; ++tj)
        acc[ti][tj] = __builtin_amdgcn_wmma_f32_16x16x32_bf16(
            false, aF[ti], false, bF[tj], (short)0, acc[ti][tj], false, false);

    if (more) {                      // fill the other LDS buffer
      if (aTransD) sstore_AT(nA, tid, va);
      else         sstore_A (nA, tid, va);
      sstore_B(nB, tid, vb);
    }
    __syncthreads();
  }

  // ---- epilogue: bias (+ residual) (+ exact GELU) ----
  const int colBase = colBlk + wc * 64;
  const bool fast = (rowBlk + BM <= M) && (colBlk + BN <= N);
  float bcol[4];
#pragma unroll
  for (int tj = 0; tj < 4; ++tj) {
    int col = colBase + tj * 16 + m16;
    bcol[tj] = bias[fast ? col : imin(col, N - 1)];
  }

  if (fast) {
#pragma unroll
    for (int ti = 0; ti < 2; ++ti)
#pragma unroll
      for (int tj = 0; tj < 4; ++tj) {
        union { v8f v; float f[8]; } cv; cv.v = acc[ti][tj];
        int col = colBase + tj * 16 + m16;
#pragma unroll
        for (int e = 0; e < 8; ++e) {
          int row = rowBlk + wr * 32 + ti * 16 + kgrp * 8 + e;  // C/D layout
          float val = cv.f[e] + bcol[tj];
          if (resid) val += resid[(size_t)row * N + col];
          if (act == 1)
            val = 0.5f * val * (1.0f + erff(val * 0.70710678118654752f));
          C[(size_t)row * N + col] = val;
        }
      }
  } else {
#pragma unroll
    for (int ti = 0; ti < 2; ++ti)
#pragma unroll
      for (int tj = 0; tj < 4; ++tj) {
        union { v8f v; float f[8]; } cv; cv.v = acc[ti][tj];
        int col = colBase + tj * 16 + m16;
#pragma unroll
        for (int e = 0; e < 8; ++e) {
          int row = rowBlk + wr * 32 + ti * 16 + kgrp * 8 + e;
          if (row < M && col < N) {
            float val = cv.f[e] + bcol[tj];
            if (resid) val += resid[(size_t)row * N + col];
            if (act == 1)
              val = 0.5f * val * (1.0f + erff(val * 0.70710678118654752f));
            C[(size_t)row * N + col] = val;
          }
        }
      }
  }
}

// ============================ LayerNorm ================================
__global__ __launch_bounds__(256) void ln_kernel(
    const float* __restrict__ x, const float* __restrict__ g,
    const float* __restrict__ be, float* __restrict__ y, int D)
{
  __shared__ float red[256];
  const int row = blockIdx.x;
  const float* xr = x + (size_t)row * D;
  float* yr = y + (size_t)row * D;
  const int tid = threadIdx.x;

  float s = 0.f;
  for (int i = tid; i < D; i += 256) s += xr[i];
  red[tid] = s; __syncthreads();
  for (int st = 128; st > 0; st >>= 1) { if (tid < st) red[tid] += red[tid + st]; __syncthreads(); }
  float mean = red[0] / (float)D;
  __syncthreads();

  float v = 0.f;
  for (int i = tid; i < D; i += 256) { float d0 = xr[i] - mean; v += d0 * d0; }
  red[tid] = v; __syncthreads();
  for (int st = 128; st > 0; st >>= 1) { if (tid < st) red[tid] += red[tid + st]; __syncthreads(); }
  float inv = rsqrtf(red[0] / (float)D + 1e-6f);

  for (int i = tid; i < D; i += 256)
    yr[i] = (xr[i] - mean) * inv * g[i] + be[i];
}

// ========================= batched transpose ===========================
// in: [B][R][C] -> out: [B][C][R]
__global__ __launch_bounds__(256) void transpose_kernel(
    const float* __restrict__ in, float* __restrict__ out, int R, int C)
{
  __shared__ float tile[32][33];
  const int b  = blockIdx.z;
  const int c0 = blockIdx.x * 32, r0 = blockIdx.y * 32;
  const float* src = in  + (size_t)b * R * C;
  float*       dst = out + (size_t)b * R * C;

  for (int i = threadIdx.y; i < 32; i += 8) {
    int r = r0 + i, c = c0 + threadIdx.x;
    tile[i][threadIdx.x] = (r < R && c < C) ? src[(size_t)r * C + c] : 0.f;
  }
  __syncthreads();
  for (int i = threadIdx.y; i < 32; i += 8) {
    int c = c0 + i, r = r0 + threadIdx.x;
    if (c < C && r < R) dst[(size_t)c * R + r] = tile[threadIdx.x][i];
  }
}

// =================== batched transpose + accumulate ====================
// out[b][c][r] += in[b][r][c]   (fused residual add for token mixing)
__global__ __launch_bounds__(256) void transpose_add_kernel(
    const float* __restrict__ in, float* __restrict__ out, int R, int C)
{
  __shared__ float tile[32][33];
  const int b  = blockIdx.z;
  const int c0 = blockIdx.x * 32, r0 = blockIdx.y * 32;
  const float* src = in  + (size_t)b * R * C;
  float*       dst = out + (size_t)b * R * C;

  for (int i = threadIdx.y; i < 32; i += 8) {
    int r = r0 + i, c = c0 + threadIdx.x;
    tile[i][threadIdx.x] = (r < R && c < C) ? src[(size_t)r * C + c] : 0.f;
  }
  __syncthreads();
  for (int i = threadIdx.y; i < 32; i += 8) {
    int c = c0 + i, r = r0 + threadIdx.x;
    if (c < C && r < R) dst[(size_t)c * R + r] += tile[threadIdx.x][i];
  }
}

// ============================ preprocess ===============================
// x:(512,2,126,128) -> xn:(512,126,256) normalized by bit-quantized scale
__global__ __launch_bounds__(256) void preprocess_kernel(
    const float* __restrict__ x, float* __restrict__ xn, float* __restrict__ sq_out)
{
  const int b = blockIdx.x;
  const float* xb = x + (size_t)b * 2 * 126 * 128;
  __shared__ float red[256];
  __shared__ float sq_sh;
  const int tid = threadIdx.x;

  float s = 0.f;
  for (int i = tid; i < 2 * 126 * 128; i += 256) { float v = xb[i] - 0.5f; s += v * v; }
  red[tid] = s; __syncthreads();
  for (int st = 128; st > 0; st >>= 1) { if (tid < st) red[tid] += red[tid + st]; __syncthreads(); }

  if (tid == 0) {
    float sv = sqrtf(red[0]);
    float sc = fminf(fmaxf(sv, 1.0f), 7.38f);
    float lg = logf(sc);
    float accum = 0.f, base = 1.f;
    for (int i = 0; i < 8; ++i) {               // FloatBiter round trip
      int bit = ((int)floorf(lg * base)) & 1;
      accum += (float)bit / base;
      base *= 2.f;
    }
    float q = expf(accum);
    sq_sh = q;
    sq_out[b] = q;
  }
  __syncthreads();

  float inv = 1.0f / sq_sh;
  for (int idx = tid; idx < 126 * 256; idx += 256) {
    int t = idx >> 8, j = idx & 255, k = j >> 1, c = j & 1;
    float v = (xb[((size_t)c * 126 + t) * 128 + k] - 0.5f) * inv;
    xn[((size_t)b * 126 + t) * 256 + j] = v;
  }
}

// ======================== VQ nearest neighbor ==========================
__global__ __launch_bounds__(256) void vq_kernel(
    const float* __restrict__ y, const float* __restrict__ embed,
    float* __restrict__ qx, int rows)
{
  __shared__ float e[256 * 8];
  __shared__ float en[256];
  for (int i = threadIdx.x; i < 256 * 8; i += 256) e[i] = embed[i];
  __syncthreads();
  {
    int i = threadIdx.x;
    float n2 = 0.f;
#pragma unroll
    for (int j = 0; j < 8; ++j) { float t = e[i * 8 + j]; n2 += t * t; }
    en[i] = n2;
  }
  __syncthreads();

  for (int r = blockIdx.x * 256 + threadIdx.x; r < rows; r += gridDim.x * 256) {
    float v[8];
#pragma unroll
    for (int j = 0; j < 8; ++j) v[j] = y[(size_t)r * 8 + j];
    float best = 3.4e38f; int bi = 0;
    for (int k = 0; k < 256; ++k) {
      float dot = 0.f;
#pragma unroll
      for (int j = 0; j < 8; ++j) dot += v[j] * e[k * 8 + j];
      float d = en[k] - 2.f * dot;               // ||y||^2 constant per row
      if (d < best) { best = d; bi = k; }
    }
#pragma unroll
    for (int j = 0; j < 8; ++j) qx[(size_t)r * 8 + j] = e[bi * 8 + j];
  }
}

// ============================ postprocess ==============================
// y:(512,126,256) -> out:(512,2,126,128); sigmoid, rescale, clip, layout.
__global__ __launch_bounds__(256) void postprocess_kernel(
    const float* __restrict__ y, const float* __restrict__ sq,
    float* __restrict__ out)
{
  const int TOT = 512 * 126 * 256;
  int idx = blockIdx.x * 256 + threadIdx.x;
  if (idx >= TOT) return;
  int b = idx / (126 * 256);
  int r = idx % (126 * 256);
  int t = r >> 8, j = r & 255, k = j >> 1, c = j & 1;
  float v  = y[idx];
  float sg = 1.0f / (1.0f + expf(-v)) - 0.5f;
  float o  = fminf(fmaxf(sg * sq[b] + 0.5f, 0.0f), 1.0f);
  out[(((size_t)b * 2 + c) * 126 + t) * 128 + k] = o;
}

// ============================= host side ===============================
namespace {

struct Lin   { const float *w, *b; };
struct Mixer {
  const float *ln1g, *ln1b, *ln2g, *ln2b;
  const float *m1f1w, *m1f1b, *m1f2w, *m1f2b;
  const float *m2f1w, *m2f1b, *m2f2w, *m2f2b;
};

inline const float* F(void* const* d_in, int& i) { return (const float*)d_in[i++]; }
inline Lin rdLin(void* const* d_in, int& i) { Lin l; l.w = F(d_in, i); l.b = F(d_in, i); return l; }
inline Mixer rdMixer(void* const* d_in, int& i) {
  Mixer m;
  m.ln1g = F(d_in, i); m.ln1b = F(d_in, i);
  m.ln2g = F(d_in, i); m.ln2b = F(d_in, i);
  m.m1f1w = F(d_in, i); m.m1f1b = F(d_in, i);
  m.m1f2w = F(d_in, i); m.m1f2b = F(d_in, i);
  m.m2f1w = F(d_in, i); m.m2f1b = F(d_in, i);
  m.m2f2w = F(d_in, i); m.m2f2b = F(d_in, i);
  return m;
}

inline void gemm(hipStream_t st, const float* A, const float* W, const float* b,
                 const float* resid, float* C, int M, int N, int K, int act,
                 int aTransD = 0) {
  dim3 grid((N + BN - 1) / BN, (M + BM - 1) / BM);
  gemm_kernel<<<grid, 256, 0, st>>>(A, W, b, resid, C, M, N, K, act, aTransD);
}

constexpr int B = 512;

inline void run_mixer(hipStream_t st, float** curp, float** altp, float* tmp,
                      float* H, const Mixer& m, int T, int D) {
  float* cur = *curp; float* alt = *altp;
  const int rows = B * T;
  dim3 tb(32, 8);
  // token mixing: x + swap(mlp1(swap(ln1(x))))
  ln_kernel<<<rows, 256, 0, st>>>(cur, m.ln1g, m.ln1b, tmp, D);
  // transposed-A GEMM reads (B,T,D) directly as (B*D, T)
  gemm(st, tmp, m.m1f1w, m.m1f1b, nullptr, H,   B * D, 4 * T, T,     1, D);
  gemm(st, H,   m.m1f2w, m.m1f2b, nullptr, alt, B * D, T,     4 * T, 0);
  // fused transpose + residual add: cur(B,T,D) += alt(B,D,T)^T
  transpose_add_kernel<<<dim3((T + 31) / 32, (D + 31) / 32, B), tb, 0, st>>>(alt, cur, D, T);
  // channel mixing: x + mlp2(ln2(x))
  ln_kernel<<<rows, 256, 0, st>>>(cur, m.ln2g, m.ln2b, tmp, D);
  gemm(st, tmp, m.m2f1w, m.m2f1b, nullptr, H,   rows, 4 * D, D,     1);
  gemm(st, H,   m.m2f2w, m.m2f2b, cur,     alt, rows, D,     4 * D, 0);
  *curp = alt; *altp = cur;
}

inline void run_chlin(hipStream_t st, float** curp, float** altp,
                      const Lin& l, int Tin, int Tout, int D) {
  dim3 tb(32, 8);
  // transposed-A GEMM: cur(B,Tin,D) read as (B*D, Tin)
  gemm(st, *curp, l.w, l.b, nullptr, *altp, B * D, Tout, Tin, 0, D);          // (B,D,Tout)
  transpose_kernel<<<dim3((Tout + 31) / 32, (D + 31) / 32, B), tb, 0, st>>>(*altp, *curp, D, Tout); // (B,Tout,D)
}

} // namespace

extern "C" void kernel_launch(void* const* d_in, const int* in_sizes, int n_in,
                              void* d_out, int out_size, void* d_ws, size_t ws_size,
                              hipStream_t stream) {
  (void)in_sizes; (void)n_in; (void)out_size; (void)ws_size;

  int i = 0;
  const float* x     = F(d_in, i);
  const float* embed = F(d_in, i);
  // encoder params
  Lin   e_fc_in  = rdLin(d_in, i);
  Mixer e_mb1    = rdMixer(d_in, i);
  Lin   e_chlin  = rdLin(d_in, i);
  Mixer e_mb2    = rdMixer(d_in, i);
  Mixer e_mb3    = rdMixer(d_in, i);
  Mixer e_mb4    = rdMixer(d_in, i);
  Lin   e_fc_out = rdLin(d_in, i);
  // decoder params
  Lin   d_fc_in  = rdLin(d_in, i);
  Mixer d_mb1    = rdMixer(d_in, i);
  Lin   d_chlin  = rdLin(d_in, i);
  Mixer d_mb2    = rdMixer(d_in, i);
  Mixer d_mb3    = rdMixer(d_in, i);
  Mixer d_mb4    = rdMixer(d_in, i);
  Mixer d_mb5    = rdMixer(d_in, i);
  Lin   d_fc_out = rdLin(d_in, i);

  // ---- workspace carve-up (floats) ----
  float* ws = (float*)d_ws;
  size_t off = 0;
  auto carve = [&](size_t n) { float* p = ws + off; off += n; return p; };
  const size_t ACT = (size_t)B * 126 * 256;      // 16.5 M floats
  float* cur = carve(ACT);
  float* alt = carve(ACT);
  float* tmp = carve(ACT);
  float* H   = carve((size_t)B * 126 * 1024);    // 66 M floats (max hidden)
  float* sq  = carve(512);
  float* Y   = carve((size_t)B * 63 * 8);
  float* QX  = carve((size_t)B * 63 * 8);

  // ---- preprocess: (b,2,126,128) -> normalized (b,126,256) + sq[b] ----
  preprocess_kernel<<<B, 256, 0, stream>>>(x, cur, sq);

  // ---- encoder ----
  gemm(stream, cur, e_fc_in.w, e_fc_in.b, nullptr, alt, B * 126, 128, 256, 0);
  { float* t0 = cur; cur = alt; alt = t0; }
  run_mixer(stream, &cur, &alt, tmp, H, e_mb1, 126, 128);
  run_chlin(stream, &cur, &alt, e_chlin, 126, 63, 128);
  run_mixer(stream, &cur, &alt, tmp, H, e_mb2, 63, 128);
  run_mixer(stream, &cur, &alt, tmp, H, e_mb3, 63, 128);
  run_mixer(stream, &cur, &alt, tmp, H, e_mb4, 63, 128);
  gemm(stream, cur, e_fc_out.w, e_fc_out.b, nullptr, Y, B * 63, 8, 128, 0);

  // ---- VQ quant/dequant ----
  vq_kernel<<<64, 256, 0, stream>>>(Y, embed, QX, B * 63);

  // ---- decoder ----
  gemm(stream, QX, d_fc_in.w, d_fc_in.b, nullptr, cur, B * 63, 256, 8, 0);
  run_mixer(stream, &cur, &alt, tmp, H, d_mb1, 63, 256);
  run_chlin(stream, &cur, &alt, d_chlin, 63, 126, 256);
  run_mixer(stream, &cur, &alt, tmp, H, d_mb2, 126, 256);
  run_mixer(stream, &cur, &alt, tmp, H, d_mb3, 126, 256);
  run_mixer(stream, &cur, &alt, tmp, H, d_mb4, 126, 256);
  run_mixer(stream, &cur, &alt, tmp, H, d_mb5, 126, 256);
  gemm(stream, cur, d_fc_out.w, d_fc_out.b, nullptr, alt, B * 126, 256, 256, 0);

  // ---- postprocess: sigmoid, rescale, clip, layout ----
  postprocess_kernel<<<(512 * 126 * 256 + 255) / 256, 256, 0, stream>>>(
      alt, sq, (float*)d_out);
}
